// DNRI_Encoder_44452911514390
// MI455X (gfx1250) — matrix operand order repaired and checked by reading, not compiled, not run on hardware
//
#include <hip/hip_runtime.h>

// ---- problem dims ----
#define Bv 8
#define Tv 64
#define Nv 24
#define Dv 6
#define Hv 128
#define Rv 64
#define NEv 4
#define Ev 552                    // N*(N-1)
#define MNODE (Bv*Nv*Tv)          // 12288
#define MEDGE (Bv*Ev*Tv)          // 282624
#define PRIOR_SZ (Bv*Tv*Ev*NEv)   // 1130496

typedef __attribute__((ext_vector_type(16))) _Float16 v16h;
typedef __attribute__((ext_vector_type(8)))  float    v8f;

__device__ __forceinline__ float elu1(float x){ return x > 0.f ? x : __expf(x) - 1.f; }
__device__ __forceinline__ float sigm(float x){ return 1.f / (1.f + __expf(-x)); }

__device__ __forceinline__ v8f wmma_f16(v16h a, v16h b, v8f c){
  // D = A(16x32 f16) * B(32x16 f16) + C(16x16 f32)
  return __builtin_amdgcn_wmma_f32_16x16x32_f16(false, a, false, b, (short)0, c, false, false);
}

// ---------------------------------------------------------------------------
// Weight convert (+optional transpose) fp32 (K,N) -> f16 (N,K) row-major
// ---------------------------------------------------------------------------
__global__ void convw_kernel(const float* __restrict__ src, _Float16* __restrict__ dst,
                             int K, int N, int transpose){
  int i = blockIdx.x * 256 + threadIdx.x;
  if (i >= K * N) return;
  if (transpose) { int n = i / K, k = i - n * K; dst[i] = (_Float16)src[(size_t)k * N + n]; }
  else           { dst[i] = (_Float16)src[i]; }
}

// ---------------------------------------------------------------------------
// MLP1 layer1: K=6 (too small for WMMA) — plain VALU GEMM with transpose-in.
// rows (b,n,t) read from inputs (B,T,N,D); out = elu(x@w1+b1), (MNODE,H)
// ---------------------------------------------------------------------------
__global__ void mlp1_l1_kernel(const float* __restrict__ inp, const float* __restrict__ w,
                               const float* __restrict__ bias, float* __restrict__ out){
  int idx = blockIdx.x * 256 + threadIdx.x;
  if (idx >= MNODE * Hv) return;
  int c   = idx & (Hv-1);
  int rem = idx >> 7;          // (b*N+n)*T + t
  int t   = rem & (Tv-1);
  int rem2= rem >> 6;          // b*N + n
  int n   = rem2 % Nv;
  int b   = rem2 / Nv;
  const float* xin = inp + ((((size_t)b * Tv + t) * Nv) + n) * Dv;
  float acc = bias[c];
  #pragma unroll
  for (int k = 0; k < Dv; ++k) acc += xin[k] * w[k * Hv + c];
  out[idx] = elu1(acc);
}

// ---------------------------------------------------------------------------
// WMMA GEMM: C[M,Nn] = act(A[M,K] @ Bt^T + bias), Bt is f16 (Nn,K) row-major.
// MODE 0: A plain fp32 (M,K)
// MODE 1: A = concat(x1[send], x1[recv])       (K=256), rows are (b,e,t)
// MODE 2: A = concat(x3[send], x3[recv], e2)   (K=384)
// Block: 256 thr = 8 waves; tile 128x64; wave tile 32x32 (2x2 WMMA).
// M%128==0, Nn%64==0, K%32==0 guaranteed by the problem dims.
// ---------------------------------------------------------------------------
template<int MODE>
__global__ __launch_bounds__(256)
void gemm_wmma_kernel(const float* __restrict__ A, const float* __restrict__ G1,
                      const float* __restrict__ G2, const _Float16* __restrict__ Bt,
                      const float* __restrict__ bias, float* __restrict__ C,
                      int M, int Nn, int K, int act)
{
  __shared__ _Float16 sA[128 * 40];   // 128 rows x 32 K, stride 40 (16B-aligned rows)
  __shared__ _Float16 sB[ 64 * 40];   // 64 N-rows x 32 K

  const int tid  = threadIdx.x;
  const int lane = tid & 31;
  const int wave = tid >> 5;
  const int m0 = blockIdx.x * 128;
  const int n0 = blockIdx.y * 64;
  const int wm = (wave & 3) * 32;
  const int wn = (wave >> 2) * 32;

  // A staging: each thread fills one 16-element half-row
  const int ar = tid >> 1;
  const int ac = (tid & 1) * 16;
  const int arow = m0 + ar;

  size_t baseA = 0, baseS = 0, baseR = 0, baseE = 0;
  if (MODE == 0) {
    baseA = (size_t)arow * K;
  } else {
    int b  = arow / (Ev * Tv);
    int rm = arow - b * (Ev * Tv);
    int ei = rm >> 6;
    int t  = rm & (Tv - 1);
    int s  = ei / (Nv - 1);
    int of = ei - s * (Nv - 1);
    int rv = of + (of >= s ? 1 : 0);
    baseS = (((size_t)b * Nv + s ) * Tv + t) * Hv;
    baseR = (((size_t)b * Nv + rv) * Tv + t) * Hv;
    baseE = (size_t)arow * Hv;
  }

  // B staging: each thread fills one 8-element chunk
  const int br = tid >> 2;
  const int bc = (tid & 3) * 8;

  v8f acc00 = {}; v8f acc01 = {}; v8f acc10 = {}; v8f acc11 = {};

  for (int k0 = 0; k0 < K; k0 += 32) {
    #pragma unroll
    for (int u = 0; u < 16; ++u) {
      int k = k0 + ac + u;
      float v;
      if (MODE == 0)       v = A[baseA + k];
      else if (MODE == 1)  v = (k < Hv) ? G1[baseS + k] : G1[baseR + (k - Hv)];
      else                 v = (k < Hv) ? G1[baseS + k]
                             : (k < 2*Hv) ? G1[baseR + (k - Hv)]
                             : G2[baseE + (k - 2*Hv)];
      sA[ar * 40 + ac + u] = (_Float16)v;
    }
    {
      const _Float16* bp = Bt + (size_t)(n0 + br) * K + (k0 + bc);
      #pragma unroll
      for (int u = 0; u < 8; ++u) sB[br * 40 + bc + u] = bp[u];
    }
    __syncthreads();

    // Build fragments in contiguous k = 16*hi + e order, same for A and B (wave32).
    const int hi = lane >> 4;    // 0/1 selects K sub-half
    const int lr = lane & 15;
    v16h a0, a1, b0, b1;
    #pragma unroll
    for (int e = 0; e < 8; ++e) {
      a0[e]     = sA[(wm      + lr) * 40 +     16*hi + e];
      a0[e + 8] = sA[(wm      + lr) * 40 + 8 + 16*hi + e];
      a1[e]     = sA[(wm + 16 + lr) * 40 +     16*hi + e];
      a1[e + 8] = sA[(wm + 16 + lr) * 40 + 8 + 16*hi + e];
    }
    #pragma unroll
    for (int e = 0; e < 16; ++e) {
      b0[e] = sB[(wn      + lr) * 40 + 16*hi + e];
      b1[e] = sB[(wn + 16 + lr) * 40 + 16*hi + e];
    }
    acc00 = wmma_f16(a0, b0, acc00);
    acc01 = wmma_f16(a0, b1, acc01);
    acc10 = wmma_f16(a1, b0, acc10);
    acc11 = wmma_f16(a1, b1, acc11);
    __syncthreads();
  }

  const int hi = lane >> 4;
  const int lr = lane & 15;
  #pragma unroll
  for (int r = 0; r < 8; ++r) {
    int row0 = m0 + wm + r + 8*hi;
    int row1 = row0 + 16;
    int col0 = n0 + wn + lr;
    int col1 = col0 + 16;
    float v00 = acc00[r] + bias[col0];
    float v01 = acc01[r] + bias[col1];
    float v10 = acc10[r] + bias[col0];
    float v11 = acc11[r] + bias[col1];
    if (act) { v00 = elu1(v00); v01 = elu1(v01); v10 = elu1(v10); v11 = elu1(v11); }
    C[(size_t)row0 * Nn + col0] = v00;
    C[(size_t)row0 * Nn + col1] = v01;
    C[(size_t)row1 * Nn + col0] = v10;
    C[(size_t)row1 * Nn + col1] = v11;
  }
}

// ---------------------------------------------------------------------------
// BatchNorm over all rows, per channel (H=128). Deterministic 2-level reduce.
// ---------------------------------------------------------------------------
__global__ void bn_stats_kernel(const float* __restrict__ x, int M, float* __restrict__ part){
  __shared__ float shs[256], shq[256];
  int tid = threadIdx.x;
  int c = tid & 127, half = tid >> 7;
  float s = 0.f, q = 0.f;
  for (int r = blockIdx.x * 2 + half; r < M; r += gridDim.x * 2) {
    float v = x[(size_t)r * Hv + c];
    s += v; q += v * v;
  }
  shs[tid] = s; shq[tid] = q;
  __syncthreads();
  if (tid < 128) {
    part[(size_t)blockIdx.x * 256 + tid]       = shs[tid] + shs[tid + 128];
    part[(size_t)blockIdx.x * 256 + 128 + tid] = shq[tid] + shq[tid + 128];
  }
}

__global__ void bn_final_kernel(const float* __restrict__ part, const float* __restrict__ g,
                                const float* __restrict__ be, int M, int nblk,
                                float* __restrict__ ss){
  int c = threadIdx.x;   // 128
  float s = 0.f, q = 0.f;
  for (int b = 0; b < nblk; ++b) { s += part[(size_t)b * 256 + c]; q += part[(size_t)b * 256 + 128 + c]; }
  float mean = s / (float)M;
  float var  = q / (float)M - mean * mean;
  float sc = g[c] * rsqrtf(var + 1e-5f);
  ss[c] = sc;
  ss[128 + c] = be[c] - mean * sc;
}

__global__ void bn_apply_kernel(float* __restrict__ x, size_t n, const float* __restrict__ ss){
  size_t idx = (size_t)blockIdx.x * 256 + threadIdx.x;
  if (idx >= n) return;
  int c = (int)(idx & 127);
  x[idx] = x[idx] * ss[c] + ss[128 + c];
}

// ---------------------------------------------------------------------------
// Edge aggregation: agg[b,n,t,:] = (1/(N-1)) * sum_{s!=n} e2[b, eid(s,n), t, :]
// ---------------------------------------------------------------------------
__global__ void agg_kernel(const float* __restrict__ e2, float* __restrict__ out){
  int idx = blockIdx.x * 256 + threadIdx.x;
  if (idx >= MNODE * Hv) return;
  int c   = idx & 127;
  int rem = idx >> 7;
  int t   = rem & 63;
  int rem2= rem >> 6;
  int n   = rem2 % Nv;
  int b   = rem2 / Nv;
  float acc = 0.f;
  #pragma unroll 4
  for (int s = 0; s < Nv; ++s) {
    if (s == n) continue;
    int of = n - (n > s ? 1 : 0);
    int ei = s * (Nv - 1) + of;
    acc += e2[((((size_t)b * Ev + ei) * Tv) + t) * Hv + c];
  }
  out[idx] = acc * (1.f / (float)(Nv - 1));
}

// ---------------------------------------------------------------------------
// GRU recurrence: one block (64 threads) per sequence; whh resident in LDS.
// xw = x@wih.T + bih precomputed by WMMA GEMM. FWD also emits hT.
// ---------------------------------------------------------------------------
template<bool FWD>
__global__ __launch_bounds__(64)
void gru_kernel(const float* __restrict__ xw, const float* __restrict__ whh,
                const float* __restrict__ bhh, float* __restrict__ y,
                float* __restrict__ hT){
  __shared__ float sW[3 * Rv * Rv];   // 192*64 fp32 = 48KB
  __shared__ float sh[Rv];
  const int seq = blockIdx.x;
  const int j = threadIdx.x;
  for (int i = j; i < 3 * Rv * Rv; i += Rv) sW[i] = whh[i];
  sh[j] = 0.f;
  __syncthreads();
  const float bh_r = bhh[j], bh_z = bhh[Rv + j], bh_n = bhh[2*Rv + j];
  const float* xwbase = xw + (size_t)seq * Tv * (3*Rv);
  float* ybase = y + (size_t)seq * Tv * Rv;
  float hnew = 0.f;
  for (int i = 0; i < Tv; ++i) {
    int t = FWD ? i : (Tv - 1 - i);
    float hr = bh_r, hz = bh_z, hn = bh_n;
    #pragma unroll 8
    for (int k = 0; k < Rv; ++k) {
      float hv = sh[k];
      hr += hv * sW[(        j) * Rv + k];
      hz += hv * sW[(Rv    + j) * Rv + k];
      hn += hv * sW[(2*Rv  + j) * Rv + k];
    }
    const float* xt = xwbase + (size_t)t * (3*Rv);
    float r = sigm(xt[j] + hr);
    float z = sigm(xt[Rv + j] + hz);
    float n = tanhf(xt[2*Rv + j] + r * hn);
    hnew = (1.f - z) * n + z * sh[j];
    __syncthreads();
    sh[j] = hnew;
    ybase[(size_t)t * Rv + j] = hnew;
    __syncthreads();
  }
  if (FWD) hT[(size_t)seq * Rv + j] = hnew;
}

// ---------------------------------------------------------------------------
// Heads: prior = fy@prior_w+b (B,T,E,NE); enc = [fy,ry]@enc_w+b (B,T,E,NE)
// ---------------------------------------------------------------------------
__global__ void head_kernel(const float* __restrict__ fy, const float* __restrict__ ry,
                            const float* __restrict__ pw, const float* __restrict__ pb,
                            const float* __restrict__ ew, const float* __restrict__ eb,
                            float* __restrict__ out){
  int idx = blockIdx.x * 256 + threadIdx.x;   // (b*T + t)*E + ei
  if (idx >= Bv * Tv * Ev) return;
  int ei  = idx % Ev;
  int rem = idx / Ev;
  int t   = rem & (Tv - 1);
  int b   = rem >> 6;
  int seq = b * Ev + ei;
  const float* f  = fy + ((size_t)seq * Tv + t) * Rv;
  const float* rr = ry + ((size_t)seq * Tv + t) * Rv;
  float p[NEv], e[NEv];
  #pragma unroll
  for (int c = 0; c < NEv; ++c) { p[c] = pb[c]; e[c] = eb[c]; }
  for (int k = 0; k < Rv; ++k) {
    float fv = f[k], rv = rr[k];
    #pragma unroll
    for (int c = 0; c < NEv; ++c) {
      p[c] += fv * pw[k * NEv + c];
      e[c] += fv * ew[k * NEv + c];
      e[c] += rv * ew[(Rv + k) * NEv + c];
    }
  }
  size_t o = (size_t)idx * NEv;
  #pragma unroll
  for (int c = 0; c < NEv; ++c) {
    out[o + c] = p[c];
    out[(size_t)PRIOR_SZ + o + c] = e[c];
  }
}

// ---------------------------------------------------------------------------
// Launch
// ---------------------------------------------------------------------------
extern "C" void kernel_launch(void* const* d_in, const int* in_sizes, int n_in,
                              void* d_out, int out_size, void* d_ws, size_t ws_size,
                              hipStream_t stream)
{
  const float* inp  = (const float*)d_in[0];
  const float* m1w1 = (const float*)d_in[1];  const float* m1b1 = (const float*)d_in[2];
  const float* m1w2 = (const float*)d_in[3];  const float* m1b2 = (const float*)d_in[4];
  const float* m1g  = (const float*)d_in[5];  const float* m1be = (const float*)d_in[6];
  const float* m2w1 = (const float*)d_in[7];  const float* m2b1 = (const float*)d_in[8];
  const float* m2w2 = (const float*)d_in[9];  const float* m2b2 = (const float*)d_in[10];
  const float* m2g  = (const float*)d_in[11]; const float* m2be = (const float*)d_in[12];
  const float* m3w1 = (const float*)d_in[13]; const float* m3b1 = (const float*)d_in[14];
  const float* m3w2 = (const float*)d_in[15]; const float* m3b2 = (const float*)d_in[16];
  const float* m3g  = (const float*)d_in[17]; const float* m3be = (const float*)d_in[18];
  const float* m4w1 = (const float*)d_in[19]; const float* m4b1 = (const float*)d_in[20];
  const float* m4w2 = (const float*)d_in[21]; const float* m4b2 = (const float*)d_in[22];
  const float* m4g  = (const float*)d_in[23]; const float* m4be = (const float*)d_in[24];
  const float* wih_f = (const float*)d_in[25]; const float* whh_f = (const float*)d_in[26];
  const float* bih_f = (const float*)d_in[27]; const float* bhh_f = (const float*)d_in[28];
  const float* wih_r = (const float*)d_in[29]; const float* whh_r = (const float*)d_in[30];
  const float* bih_r = (const float*)d_in[31]; const float* bhh_r = (const float*)d_in[32];
  const float* prior_w = (const float*)d_in[33]; const float* prior_b = (const float*)d_in[34];
  const float* enc_w   = (const float*)d_in[35]; const float* enc_b   = (const float*)d_in[36];

  // ---- workspace carve-up (f16 weights first, then fp32 tensors) ----
  const size_t O_M1W2 = 0;
  const size_t O_M2W1 = O_M1W2 + 128*128;
  const size_t O_M2W2 = O_M2W1 + 128*256;
  const size_t O_M3W1 = O_M2W2 + 128*128;
  const size_t O_M3W2 = O_M3W1 + 128*128;
  const size_t O_M4W1 = O_M3W2 + 128*128;
  const size_t O_M4W2 = O_M4W1 + 128*384;
  const size_t O_WIHF = O_M4W2 + 128*128;
  const size_t O_WIHR = O_WIHF + 192*128;
  const size_t F16_TOTAL = O_WIHR + 192*128;     // 212992 halves
  _Float16* wf16 = (_Float16*)d_ws;
  float* fws = (float*)((char*)d_ws + ((F16_TOTAL * 2 + 255) & ~((size_t)255)));

  float* x1    = fws;                               // MNODE*H
  float* hnode = x1    + (size_t)MNODE * Hv;        // MNODE*H
  float* aggb  = hnode + (size_t)MNODE * Hv;        // MNODE*H
  float* x3    = aggb  + (size_t)MNODE * Hv;        // MNODE*H
  float* ss    = x3    + (size_t)MNODE * Hv;        // 256
  float* part  = ss + 256;                          // 256 blocks * 256
  float* hbig  = part + 256 * 256;                  // MEDGE*H
  float* e2    = hbig + (size_t)MEDGE * Hv;         // MEDGE*H
  float* e4    = e2   + (size_t)MEDGE * Hv;         // MEDGE*H
  float* xwf   = e4   + (size_t)MEDGE * Hv;         // MEDGE*192
  float* xwr   = xwf  + (size_t)MEDGE * 192;        // MEDGE*192
  float* fy    = xwr  + (size_t)MEDGE * 192;        // MEDGE*R
  float* ry    = fy   + (size_t)MEDGE * Rv;         // MEDGE*R
  float* outp  = (float*)d_out;
  float* hT    = outp + 2 * (size_t)PRIOR_SZ;

  auto cv = [&](const float* src, _Float16* dst, int K, int N, int tr){
    int n = K * N;
    convw_kernel<<<(n + 255) / 256, 256, 0, stream>>>(src, dst, K, N, tr);
  };
  cv(m1w2, wf16 + O_M1W2, 128, 128, 1);
  cv(m2w1, wf16 + O_M2W1, 256, 128, 1);
  cv(m2w2, wf16 + O_M2W2, 128, 128, 1);
  cv(m3w1, wf16 + O_M3W1, 128, 128, 1);
  cv(m3w2, wf16 + O_M3W2, 128, 128, 1);
  cv(m4w1, wf16 + O_M4W1, 384, 128, 1);
  cv(m4w2, wf16 + O_M4W2, 128, 128, 1);
  cv(wih_f, wf16 + O_WIHF, 192, 128, 0);   // wih is (3R,H) == Bt(N,K) already
  cv(wih_r, wf16 + O_WIHR, 192, 128, 0);

  auto bn = [&](float* x, int M, const float* g, const float* be){
    bn_stats_kernel<<<256, 256, 0, stream>>>(x, M, part);
    bn_final_kernel<<<1, 128, 0, stream>>>(part, g, be, M, 256, ss);
    size_t n = (size_t)M * Hv;
    bn_apply_kernel<<<(unsigned)((n + 255) / 256), 256, 0, stream>>>(x, n, ss);
  };

  // MLP1: (B,N,T,D) -> x1
  mlp1_l1_kernel<<<(MNODE * Hv + 255) / 256, 256, 0, stream>>>(inp, m1w1, m1b1, hnode);
  gemm_wmma_kernel<0><<<dim3(MNODE/128, Hv/64), 256, 0, stream>>>(
      hnode, nullptr, nullptr, wf16 + O_M1W2, m1b2, x1, MNODE, Hv, 128, 1);
  bn(x1, MNODE, m1g, m1be);

  // MLP2 (gathered concat of x1 over send/recv): -> e2
  gemm_wmma_kernel<1><<<dim3(MEDGE/128, Hv/64), 256, 0, stream>>>(
      nullptr, x1, nullptr, wf16 + O_M2W1, m2b1, hbig, MEDGE, Hv, 256, 1);
  gemm_wmma_kernel<0><<<dim3(MEDGE/128, Hv/64), 256, 0, stream>>>(
      hbig, nullptr, nullptr, wf16 + O_M2W2, m2b2, e2, MEDGE, Hv, 128, 1);
  bn(e2, MEDGE, m2g, m2be);

  // segment-sum over RECV, /(N-1)
  agg_kernel<<<(MNODE * Hv + 255) / 256, 256, 0, stream>>>(e2, aggb);

  // MLP3: aggb -> x3
  gemm_wmma_kernel<0><<<dim3(MNODE/128, Hv/64), 256, 0, stream>>>(
      aggb, nullptr, nullptr, wf16 + O_M3W1, m3b1, hnode, MNODE, Hv, 128, 1);
  gemm_wmma_kernel<0><<<dim3(MNODE/128, Hv/64), 256, 0, stream>>>(
      hnode, nullptr, nullptr, wf16 + O_M3W2, m3b2, x3, MNODE, Hv, 128, 1);
  bn(x3, MNODE, m3g, m3be);

  // MLP4 (gathered concat x3[send],x3[recv],e2): -> e4 (= GRU input sequence)
  gemm_wmma_kernel<2><<<dim3(MEDGE/128, Hv/64), 256, 0, stream>>>(
      nullptr, x3, e2, wf16 + O_M4W1, m4b1, hbig, MEDGE, Hv, 384, 1);
  gemm_wmma_kernel<0><<<dim3(MEDGE/128, Hv/64), 256, 0, stream>>>(
      hbig, nullptr, nullptr, wf16 + O_M4W2, m4b2, e4, MEDGE, Hv, 128, 1);
  bn(e4, MEDGE, m4g, m4be);

  // GRU input projections: xw = seq @ wih.T + bih   (N=192, K=128)
  gemm_wmma_kernel<0><<<dim3(MEDGE/128, 192/64), 256, 0, stream>>>(
      e4, nullptr, nullptr, wf16 + O_WIHF, bih_f, xwf, MEDGE, 192, 128, 0);
  gemm_wmma_kernel<0><<<dim3(MEDGE/128, 192/64), 256, 0, stream>>>(
      e4, nullptr, nullptr, wf16 + O_WIHR, bih_r, xwr, MEDGE, 192, 128, 0);

  // GRU recurrences (4416 sequences, T=64)
  gru_kernel<true ><<<Bv * Ev, Rv, 0, stream>>>(xwf, whh_f, bhh_f, fy, hT);
  gru_kernel<false><<<Bv * Ev, Rv, 0, stream>>>(xwr, whh_r, bhh_r, ry, nullptr);

  // Output heads
  head_kernel<<<(Bv * Tv * Ev + 255) / 256, 256, 0, stream>>>(
      fy, ry, prior_w, prior_b, enc_w, enc_b, outp);
}
